// StandardAttention_1348619731139
// MI455X (gfx1250) — matrix operand order repaired and checked by reading, not compiled, not run on hardware
//
#include <hip/hip_runtime.h>
#include <cstdint>

#define D_MODEL  1024
#define N_HEAD   16
#define HEAD_DIM 64
#define BATCH    2
#define SEQ      2048
#define MTOT     (BATCH * SEQ)

typedef _Float16 v16h __attribute__((ext_vector_type(16)));
typedef _Float16 v8h  __attribute__((ext_vector_type(8)));
typedef float    v8f  __attribute__((ext_vector_type(8)));

// Load two contiguous 8-half chunks (16B each) into one 16-half fragment.
static __device__ __forceinline__ v16h ld_frag2(const _Float16* p0, const _Float16* p1) {
    v8h a = *(const v8h*)p0;
    v8h b = *(const v8h*)p1;
    v16h r;
#pragma unroll
    for (int i = 0; i < 8; ++i) { r[i] = a[i]; r[i + 8] = b[i]; }
    return r;
}

// CDNA5 async global->LDS copy: 16B per lane, tracked by ASYNCcnt.
static __device__ __forceinline__ void async_copy_b128(uint32_t lds_off, const void* gaddr) {
    asm volatile("global_load_async_to_lds_b128 %0, %1, off"
                 :: "v"(lds_off), "v"((uint64_t)(uintptr_t)gaddr)
                 : "memory");
}
static __device__ __forceinline__ void wait_asynccnt0() {
    asm volatile("s_wait_asynccnt 0" ::: "memory");
}

__global__ void cvt_f32_to_f16(const float* __restrict__ in, _Float16* __restrict__ out, int n) {
    int i = blockIdx.x * blockDim.x + threadIdx.x;
    if (i < n) out[i] = (_Float16)in[i];
}

// C[M,N] = A[M,K] (f16) @ Bw[N,K]^T (f16), f32 accumulate via v_wmma_f32_16x16x32_f16.
// Block = 8 waves covering 256(M) x 64(N); each wave a 32x64 strip.
// The block-shared B tile (64 cols x 64 K, 8KB) is staged in LDS with double-buffered
// global_load_async_to_lds_b128 DMA: 16 WMMAs per wave per barrier, DMA of the next
// stage overlapping compute of the current one.
// MODE 0: f32 row-major output [M,N]
// MODE 1: RoPE fused epilogue, f16 output laid out [B,H,T,hd]   (Q and K)
// MODE 2: f16 output transposed per head [B,H,hd,T]             (V)
template <int MODE>
__global__ __launch_bounds__(256) void gemm_wmma(
    const _Float16* __restrict__ A, const _Float16* __restrict__ Bw,
    float* __restrict__ outF, _Float16* __restrict__ outH,
    const float* __restrict__ freqs, int M, int N, int K)
{
    __shared__ _Float16 bstage[2][64 * 64];   // [buf][col*64 + k], 8KB each

    const int tid   = threadIdx.x;
    const int lane  = tid & 31;
    const int wave  = tid >> 5;
    const int n15   = lane & 15;
    const int hi    = lane >> 4;          // 0 = lanes 0-15, 1 = lanes 16-31
    const int mtile = blockIdx.x * 256 + wave * 32;
    const int ntile = blockIdx.y * 64;

    // Async-copy assignment: thread tid moves 2 x 16B of the 8KB B tile.
    const int ccol = tid >> 2;            // 0..63
    const int cq   = tid & 3;             // chunk pair selector

    v8f zero = {};
    v8f acc[2][4];
#pragma unroll
    for (int s = 0; s < 2; ++s)
#pragma unroll
        for (int j = 0; j < 4; ++j) acc[s][j] = zero;

    const _Float16* browg = Bw + (size_t)(ntile + ccol) * K;  // this thread's B row
    const uint32_t  lrow0 = (uint32_t)(uintptr_t)(&bstage[0][ccol * 64]);
    const uint32_t  lrow1 = (uint32_t)(uintptr_t)(&bstage[1][ccol * 64]);

    // Prologue: stage first B tile (K 0..63).
    async_copy_b128(lrow0 + (uint32_t)(cq * 16),        browg + cq * 8);
    async_copy_b128(lrow0 + (uint32_t)((cq + 4) * 16),  browg + (cq + 4) * 8);

    int buf = 0;
    for (int k0 = 0; k0 < K; k0 += 64, buf ^= 1) {
        wait_asynccnt0();       // this wave's portion of bstage[buf] has landed
        __syncthreads();        // everyone's portion landed / prev reads done

        if (k0 + 64 < K) {      // kick off DMA for the next stage into the other buffer
            const uint32_t lr = buf ? lrow0 : lrow1;
            const _Float16* g = browg + (k0 + 64);
            async_copy_b128(lr + (uint32_t)(cq * 16),       g + cq * 8);
            async_copy_b128(lr + (uint32_t)((cq + 4) * 16), g + (cq + 4) * 8);
        }

        // Prefetch next A stage (speculative; safe past end). One row per lane.
        __builtin_prefetch(A + (size_t)(mtile + lane) * K + k0 + 64, 0, 1);

        // A fragments for both K=32 sub-steps (16x32 each):
        // lane = row, low lanes hold K [0..7|16..23], high [8..15|24..31].
        const _Float16* ap0 = A + (size_t)(mtile + n15) * K + k0 + (hi ? 8 : 0);
        const _Float16* ap1 = ap0 + (size_t)16 * K;
        v16h af[2][2];
#pragma unroll
        for (int kk = 0; kk < 2; ++kk) {
            af[0][kk] = ld_frag2(ap0 + kk * 32, ap0 + kk * 32 + 16);
            af[1][kk] = ld_frag2(ap1 + kk * 32, ap1 + kk * 32 + 16);
        }

        const _Float16* bb = &bstage[buf][0];
#pragma unroll
        for (int kk = 0; kk < 2; ++kk) {
#pragma unroll
            for (int j = 0; j < 4; ++j) {
                // B fragment (32x16) from LDS: lane = col; low lanes K [0..7|16..23], high [8..15|24..31].
                const _Float16* bp = bb + (j * 16 + n15) * 64 + kk * 32 + (hi ? 8 : 0);
                v16h bf = ld_frag2(bp, bp + 16);
                acc[0][j] = __builtin_amdgcn_wmma_f32_16x16x32_f16(
                    false, af[0][kk], false, bf, (short)0, acc[0][j], false, false);
                acc[1][j] = __builtin_amdgcn_wmma_f32_16x16x32_f16(
                    false, af[1][kk], false, bf, (short)0, acc[1][j], false, false);
            }
        }
    }

#pragma unroll
    for (int s = 0; s < 2; ++s) {
#pragma unroll
        for (int j = 0; j < 4; ++j) {
#pragma unroll
            for (int r = 0; r < 8; ++r) {
                const int row = mtile + s * 16 + r + hi * 8;  // global M index (b*T + t)
                const int col = ntile + j * 16 + n15;         // global N index (h*64 + d)
                const float v = acc[s][j][r];
                if constexpr (MODE == 0) {
                    outF[(size_t)row * N + col] = v;
                } else if constexpr (MODE == 1) {
                    // RoPE: pair partner (d^1) is in lane^1 of the same half.
                    float partner = __shfl_xor(v, 1, 32);
                    const int d = col & (HEAD_DIM - 1);
                    const int h = col >> 6;
                    const int t = row & (SEQ - 1);
                    const int b = row >> 11;                  // row / SEQ
                    const float c  = freqs[t * HEAD_DIM + (d & ~1)];
                    const float sn = freqs[t * HEAD_DIM + (d & ~1) + 1];
                    const float o = (d & 1) ? (v * c + partner * sn)   // im = xe*s + xo*c
                                            : (v * c - partner * sn);  // re = xe*c - xo*s
                    outH[(((size_t)(b * N_HEAD + h)) * SEQ + t) * HEAD_DIM + d] = (_Float16)o;
                } else {
                    const int d = col & (HEAD_DIM - 1);
                    const int h = col >> 6;
                    const int t = row & (SEQ - 1);
                    const int b = row >> 11;
                    outH[(((size_t)(b * N_HEAD + h)) * HEAD_DIM + d) * SEQ + t] = (_Float16)v;
                }
            }
        }
    }
}

// One wave per (b,h,16-query tile). Online-softmax flash attention, WMMA everywhere.
// Row sums come from an extra WMMA against a ones-matrix (C-layout replicates the row
// sum across all 16 lanes of each half) instead of ds_bpermute butterflies.
// qh/kh: [B,H,T,64] f16 (RoPE applied). vth: [B,H,64,T] f16 (transposed). ah: [B*T,1024] f16.
__global__ __launch_bounds__(32) void flash_attn(
    const _Float16* __restrict__ qh, const _Float16* __restrict__ kh,
    const _Float16* __restrict__ vth, _Float16* __restrict__ ah)
{
    __shared__ _Float16 plds[16 * 32];   // P tile bounce buffer: C-layout -> A-layout
    const int lane  = threadIdx.x;
    const int n15   = lane & 15;
    const int hi    = lane >> 4;
    const int qt    = blockIdx.x;
    const int bh    = blockIdx.y;
    const int b     = bh >> 4;
    const int h     = bh & 15;
    const int qbase = qt * 16;

    const _Float16* qp = qh  + (size_t)bh * SEQ * HEAD_DIM;
    const _Float16* kp = kh  + (size_t)bh * SEQ * HEAD_DIM;
    const _Float16* vp = vth + (size_t)bh * HEAD_DIM * SEQ;

    // Q tile as two A fragments (hd 0-31, hd 32-63), loop-invariant.
    const _Float16* aqp = qp + (size_t)(qbase + n15) * HEAD_DIM + (hi ? 16 : 0);
    const v16h aq0 = ld_frag2(aqp,      aqp + 8);
    const v16h aq1 = ld_frag2(aqp + 32, aqp + 40);

    v16h ones;
#pragma unroll
    for (int i = 0; i < 16; ++i) ones[i] = (_Float16)1.0f;

    v8f zero = {};
    v8f o[4];
    float mrow[8], lrow[8];
#pragma unroll
    for (int j = 0; j < 4; ++j) o[j] = zero;
#pragma unroll
    for (int r = 0; r < 8; ++r) { mrow[r] = -1e30f; lrow[r] = 0.0f; }

    const float SC = 0.125f * 1.44269504088896340736f;  // hd^-0.5 * log2(e)
    const int nkb = (qbase + 15) / 32 + 1;              // causal: key blocks needed

    for (int kb = 0; kb < nkb; ++kb) {
        const int kbase = kb * 32;

        if (kb + 1 < nkb) {
            // Prefetch next key block: K rows (one 128B line per lane) and V^T chunks.
            __builtin_prefetch(kp + (size_t)(kbase + 32 + lane) * HEAD_DIM, 0, 1);
            __builtin_prefetch(vp + (size_t)lane * SEQ + kbase + 32, 0, 1);
            __builtin_prefetch(vp + (size_t)(32 + lane) * SEQ + kbase + 32, 0, 1);
        }

        // S = Q @ K^T for 32 keys: two 16-key N tiles, each accumulated over hd in two K=32 chunks.
        v8f s0 = zero, s1 = zero;
        {
            const _Float16* k0p = kp + (size_t)(kbase + n15) * HEAD_DIM + (hi ? 16 : 0);
            v16h b0lo = ld_frag2(k0p,      k0p + 8);
            v16h b0hi = ld_frag2(k0p + 32, k0p + 40);
            s0 = __builtin_amdgcn_wmma_f32_16x16x32_f16(false, aq0, false, b0lo, (short)0, s0, false, false);
            s0 = __builtin_amdgcn_wmma_f32_16x16x32_f16(false, aq1, false, b0hi, (short)0, s0, false, false);
            const _Float16* k1p = k0p + 16 * HEAD_DIM;
            v16h b1lo = ld_frag2(k1p,      k1p + 8);
            v16h b1hi = ld_frag2(k1p + 32, k1p + 40);
            s1 = __builtin_amdgcn_wmma_f32_16x16x32_f16(false, aq0, false, b1lo, (short)0, s1, false, false);
            s1 = __builtin_amdgcn_wmma_f32_16x16x32_f16(false, aq1, false, b1hi, (short)0, s1, false, false);
        }

        const bool domask = (kbase + 31 > qbase);
        float fscale[8];
#pragma unroll
        for (int r = 0; r < 8; ++r) {
            float z0 = s0[r] * SC;
            float z1 = s1[r] * SC;
            if (domask) {
                const int qi = qbase + r + hi * 8;
                if (kbase + n15 > qi)      z0 = -1e30f;
                if (kbase + 16 + n15 > qi) z1 = -1e30f;
            }
            // Row max lives across the 16 lanes of this half-wave: butterfly reduce.
            float t = fmaxf(z0, z1);
            t = fmaxf(t, __shfl_xor(t, 1, 32));
            t = fmaxf(t, __shfl_xor(t, 2, 32));
            t = fmaxf(t, __shfl_xor(t, 4, 32));
            t = fmaxf(t, __shfl_xor(t, 8, 32));
            const float mn = fmaxf(mrow[r], t);
            fscale[r] = exp2f(mrow[r] - mn);
            mrow[r] = mn;
            const float p0 = exp2f(z0 - mn);
            const float p1 = exp2f(z1 - mn);
            // Stash P (C-layout element (m = r+8*hi, k = tile*16+n15)) as f16 in LDS.
            plds[(r + hi * 8) * 32 + n15]      = (_Float16)p0;
            plds[(r + hi * 8) * 32 + 16 + n15] = (_Float16)p1;
        }
#pragma unroll
        for (int j = 0; j < 4; ++j)
#pragma unroll
            for (int r = 0; r < 8; ++r) o[j][r] *= fscale[r];

        __syncthreads();
        // Re-read P in A-fragment layout: lane = row, K chunks per half.
        const _Float16* pp = plds + n15 * 32 + (hi ? 16 : 0);
        const v16h apf = ld_frag2(pp, pp + 8);

        // Row sums via the matrix pipe: P @ ones -> every lane holds its row's sum.
        v8f lsum = __builtin_amdgcn_wmma_f32_16x16x32_f16(
            false, apf, false, ones, (short)0, zero, false, false);
#pragma unroll
        for (int r = 0; r < 8; ++r) lrow[r] = lrow[r] * fscale[r] + lsum[r];

        // O += P @ V : V^T layout makes each B fragment two contiguous 16B loads.
#pragma unroll
        for (int j = 0; j < 4; ++j) {
            const _Float16* vptr = vp + (size_t)(j * 16 + n15) * SEQ + kbase + (hi ? 16 : 0);
            v16h bv = ld_frag2(vptr, vptr + 8);
            o[j] = __builtin_amdgcn_wmma_f32_16x16x32_f16(false, apf, false, bv, (short)0, o[j], false, false);
        }
        __syncthreads();
    }

    // Normalize and store f16 at [b*T + q, h*64 + d] for the output projection GEMM.
#pragma unroll
    for (int r = 0; r < 8; ++r) {
        const float inv = 1.0f / lrow[r];
        const int row = b * SEQ + qbase + r + hi * 8;
#pragma unroll
        for (int j = 0; j < 4; ++j) {
            const float val = o[j][r] * inv;
            ah[(size_t)row * D_MODEL + h * HEAD_DIM + j * 16 + n15] = (_Float16)val;
        }
    }
}

extern "C" void kernel_launch(void* const* d_in, const int* in_sizes, int n_in,
                              void* d_out, int out_size, void* d_ws, size_t ws_size,
                              hipStream_t stream)
{
    const float* x  = (const float*)d_in[0];
    const float* fr = (const float*)d_in[1];   // freqs_cis [T, 32, 2]
    const float* Wq = (const float*)d_in[2];
    const float* Wk = (const float*)d_in[3];
    const float* Wv = (const float*)d_in[4];
    const float* Wo = (const float*)d_in[5];

    const size_t NX = (size_t)MTOT * D_MODEL;     // 4M halfs
    const size_t NW = (size_t)D_MODEL * D_MODEL;  // 1M halfs

    _Float16* w   = (_Float16*)d_ws;
    _Float16* xh  = w;  w += NX;
    _Float16* wqh = w;  w += NW;
    _Float16* wkh = w;  w += NW;
    _Float16* wvh = w;  w += NW;
    _Float16* woh = w;  w += NW;
    _Float16* qh  = w;  w += NX;
    _Float16* kh  = w;  w += NX;
    _Float16* vth = w;  w += NX;
    _Float16* ah  = w;  w += NX;
    (void)ws_size; (void)in_sizes; (void)n_in; (void)out_size;

    const int thr = 256;
    cvt_f32_to_f16<<<(int)((NX + thr - 1) / thr), thr, 0, stream>>>(x,  xh,  (int)NX);
    cvt_f32_to_f16<<<(int)((NW + thr - 1) / thr), thr, 0, stream>>>(Wq, wqh, (int)NW);
    cvt_f32_to_f16<<<(int)((NW + thr - 1) / thr), thr, 0, stream>>>(Wk, wkh, (int)NW);
    cvt_f32_to_f16<<<(int)((NW + thr - 1) / thr), thr, 0, stream>>>(Wv, wvh, (int)NW);
    cvt_f32_to_f16<<<(int)((NW + thr - 1) / thr), thr, 0, stream>>>(Wo, woh, (int)NW);

    dim3 ggrid(MTOT / 256, D_MODEL / 64);
    gemm_wmma<1><<<ggrid, 256, 0, stream>>>(xh, wqh, nullptr, qh,  fr,      MTOT, D_MODEL, D_MODEL);
    gemm_wmma<1><<<ggrid, 256, 0, stream>>>(xh, wkh, nullptr, kh,  fr,      MTOT, D_MODEL, D_MODEL);
    gemm_wmma<2><<<ggrid, 256, 0, stream>>>(xh, wvh, nullptr, vth, nullptr, MTOT, D_MODEL, D_MODEL);

    flash_attn<<<dim3(SEQ / 16, BATCH * N_HEAD), 32, 0, stream>>>(qh, kh, vth, ah);

    gemm_wmma<0><<<ggrid, 256, 0, stream>>>(ah, woh, (float*)d_out, nullptr, nullptr,
                                            MTOT, D_MODEL, D_MODEL);
}
